// ActioinExpertAttention_77335181132511
// MI455X (gfx1250) — matrix operand order repaired and checked by reading, not compiled, not run on hardware
//
#include <hip/hip_runtime.h>
#include <stdint.h>

// ---------------------------------------------------------------------------
// MI455X (gfx1250) action-expert attention, bf16 WMMA pipeline + TDM staging.
//  B=16 Q=128 VLM=2048 H=16 HKV=8 D=128 HID=2048 G=2 KV=2176
// ---------------------------------------------------------------------------

#define B_    16
#define Q_    128
#define VLM_  2048
#define H_    16
#define HKV_  8
#define D_    128
#define HID_  2048
#define G_    2
#define KV_   (VLM_ + Q_)            // 2176
#define NQKV_ ((H_ + 2 * HKV_) * D_) // 4096

typedef __attribute__((ext_vector_type(16))) __bf16        v16bf;
typedef __attribute__((ext_vector_type(8)))  float         v8f;
typedef __attribute__((ext_vector_type(4)))  unsigned int  v4u;
typedef __attribute__((ext_vector_type(8)))  int           v8i;
typedef __attribute__((ext_vector_type(4)))  int           v4i;

// ---- bf16 <-> f32 (round-to-nearest-even), raw ushort storage -------------
__device__ __forceinline__ unsigned short f2bf(float f) {
  union { float f; unsigned u; } c; c.f = f;
  unsigned u = c.u;
  unsigned r = 0x7FFFu + ((u >> 16) & 1u);
  return (unsigned short)((u + r) >> 16);
}
__device__ __forceinline__ float bf2f(unsigned short h) {
  union { unsigned u; float f; } c; c.u = ((unsigned)h) << 16; return c.f;
}

// ---- WMMA wrapper ----------------------------------------------------------
__device__ __forceinline__ v8f wmma_bf16(v16bf a, v16bf b, v8f c) {
  return __builtin_amdgcn_wmma_f32_16x16x32_bf16(
      /*neg_a=*/false, a, /*neg_b=*/false, b,
      /*c_mod=*/(short)0, c, /*reuse_a=*/false, /*reuse_b=*/false);
}

// ---- Fragment loaders (contiguous k = 16*half + s, same in A and B) ----------
// A: 16x32 bf16, row-major source. Lane l: m=l%16, half=l/16;
//   dword j(0..7) = K(half*16+2j,+1).
__device__ __forceinline__ v16bf load_a_frag(const unsigned short* A, int lda,
                                             int m0, int k0, int lane) {
  const int half = lane >> 4, mr = lane & 15;
  const uint32_t* row =
      reinterpret_cast<const uint32_t*>(A + (size_t)(m0 + mr) * lda + k0);
  union { uint32_t u[8]; v16bf v; } f;
#pragma unroll
  for (int j = 0; j < 8; ++j) {
    f.u[j]     = row[half * 8 + j];
  }
  return f.v;
}

// B: 32x16 bf16 where source W is (N x K) row-major (B = W^T tile).
//   Lane l: n=l%16, K-range = (l/16)*16..+15, dword j = K(kbase+2j,+1).
__device__ __forceinline__ v16bf load_b_frag(const unsigned short* W, int ldb,
                                             int n0, int k0, int lane) {
  const int half = lane >> 4, nr = lane & 15;
  const uint32_t* row =
      reinterpret_cast<const uint32_t*>(W + (size_t)(n0 + nr) * ldb + k0);
  union { uint32_t u[8]; v16bf v; } f;
#pragma unroll
  for (int j = 0; j < 8; ++j) f.u[j] = row[half * 8 + j];
  return f.v;
}

// B: 32x16 bf16 where source V is (K x N) row-major (used for P·V from LDS).
__device__ __forceinline__ v16bf load_b_frag_kn(const unsigned short* Vm,
                                                int ldn, int k0, int n0,
                                                int lane) {
  const int half = lane >> 4, nr = lane & 15;
  union { unsigned short h[16]; v16bf v; } f;
#pragma unroll
  for (int j = 0; j < 16; ++j)
    f.h[j] = Vm[(size_t)(k0 + half * 16 + j) * ldn + n0 + nr];
  return f.v;
}

// ---- TDM: DMA a 32x128 bf16 tile (row-major, stride D_) into LDS ----------
// D# per CDNA5 ISA (2D tensor, groups 2/3 zero).
// 6-arg builtin form (clang-23 / therock-10.0 headers).
__device__ __forceinline__ void tdm_load_tile_32x128(const unsigned short* gsrc,
                                                     unsigned int lds_byte) {
  unsigned long long ga = (unsigned long long)(uintptr_t)gsrc;
  v4u g0;
  g0.x = 1u;                                  // count=1 (valid user D#)
  g0.y = lds_byte;                            // lds_addr
  g0.z = (unsigned int)(ga & 0xFFFFFFFFu);    // global_addr[31:0]
  g0.w = (unsigned int)((ga >> 32) & 0x1FFFFFFu) | (2u << 30);  // [56:32]|type=2
  v8i g1;
  g1[0] = 0x00010000;          // wg_mask=0, data_size=1 (2B), no pad/iterate
  g1[1] = (int)(128u << 16);   // tensor_dim0 = 128 elements  ([79:48] lo16)
  g1[2] = (int)(32u << 16);    // tensor_dim0 hi=0 | tensor_dim1 = 32 (lo16)
  g1[3] = (int)(128u << 16);   // tensor_dim1 hi=0 | tile_dim0 = 128
  g1[4] = 32;                  // tile_dim1 = 32, tile_dim2 = 0
  g1[5] = 128;                 // tensor_dim0_stride = 128 elements
  g1[6] = 0;                   // stride0 hi | tensor_dim1_stride lo
  g1[7] = 0;
  v4i z4 = {0, 0, 0, 0};
  v8i z8 = {0, 0, 0, 0, 0, 0, 0, 0};
  __builtin_amdgcn_tensor_load_to_lds(g0, g1, z4, z4, z8, 0);
}

// ---------------------------------------------------------------------------
// Kernel 1: elementwise f32 -> bf16
// ---------------------------------------------------------------------------
__global__ void k_f32_to_bf16(const float* __restrict__ in,
                              unsigned short* __restrict__ out, int n) {
  int i = blockIdx.x * blockDim.x + threadIdx.x;
  int stride = gridDim.x * blockDim.x;
  for (; i < n; i += stride) out[i] = f2bf(in[i]);
}

// ---------------------------------------------------------------------------
// Kernel 2: RoPE + convert the VLM prefix of K, convert V, into bf16 caches.
// ---------------------------------------------------------------------------
__global__ void k_rope_vlm(const float* __restrict__ vk,
                           const float* __restrict__ vv,
                           unsigned short* __restrict__ Kc,
                           unsigned short* __restrict__ Vc) {
  size_t idx = (size_t)blockIdx.x * blockDim.x + threadIdx.x;
  const size_t total = (size_t)B_ * HKV_ * VLM_ * 64;
  if (idx >= total) return;
  int d0 = (int)(idx & 63);
  size_t r = idx >> 6;
  int t  = (int)(r % VLM_); r /= VLM_;
  int kh = (int)(r % HKV_);
  int b  = (int)(r / HKV_);

  float f = __powf(10000.0f, -(float)d0 * (1.0f / 64.0f));
  float sn, cs;
  __sincosf((float)t * f, &sn, &cs);

  size_t src = (((size_t)(b * HKV_ + kh) * VLM_) + t) * D_ + d0;
  size_t dst = (((size_t)(b * HKV_ + kh) * KV_)  + t) * D_ + d0;
  float x0 = vk[src], x1 = vk[src + 64];
  Kc[dst]      = f2bf(x0 * cs - x1 * sn);
  Kc[dst + 64] = f2bf(x1 * cs + x0 * sn);
  Vc[dst]      = f2bf(vv[src]);
  Vc[dst + 64] = f2bf(vv[src + 64]);
}

// ---------------------------------------------------------------------------
// Kernel 3: C = A (MxK) · W^T, W given (NxK), bf16 in / f32 accum.
// 8 waves (2x4) per block; each wave owns 32x32; K-loop software pipelined.
// ---------------------------------------------------------------------------
template <bool F32OUT>
__global__ void __launch_bounds__(256) k_gemm_bf16_nt(
    const unsigned short* __restrict__ A, const unsigned short* __restrict__ W,
    float* __restrict__ Cf, unsigned short* __restrict__ Cb, int M, int N,
    int K) {
  const int lane = threadIdx.x & 31;
  const int wave = threadIdx.x >> 5;
  const int m0 = blockIdx.x * 64 + (wave >> 2) * 32;
  const int n0 = blockIdx.y * 128 + (wave & 3) * 32;
  if (m0 >= M || n0 >= N) return;

  v8f c00 = {}, c01 = {}, c10 = {}, c11 = {};
  v16bf a0 = load_a_frag(A, K, m0, 0, lane);
  v16bf a1 = load_a_frag(A, K, m0 + 16, 0, lane);
  v16bf b0 = load_b_frag(W, K, n0, 0, lane);
  v16bf b1 = load_b_frag(W, K, n0 + 16, 0, lane);
  for (int k0 = 0; k0 < K; k0 += 32) {
    v16bf na0, na1, nb0, nb1;
    if (k0 + 32 < K) {  // prefetch next K-slice while WMMAs drain
      na0 = load_a_frag(A, K, m0, k0 + 32, lane);
      na1 = load_a_frag(A, K, m0 + 16, k0 + 32, lane);
      nb0 = load_b_frag(W, K, n0, k0 + 32, lane);
      nb1 = load_b_frag(W, K, n0 + 16, k0 + 32, lane);
    }
    c00 = wmma_bf16(a0, b0, c00);
    c01 = wmma_bf16(a0, b1, c01);
    c10 = wmma_bf16(a1, b0, c10);
    c11 = wmma_bf16(a1, b1, c11);
    a0 = na0; a1 = na1; b0 = nb0; b1 = nb1;
  }

  const int half = lane >> 4, nr = lane & 15;
#pragma unroll
  for (int mt = 0; mt < 2; ++mt) {
#pragma unroll
    for (int nt = 0; nt < 2; ++nt) {
      v8f c = mt == 0 ? (nt == 0 ? c00 : c01) : (nt == 0 ? c10 : c11);
      int mb = m0 + mt * 16 + half * 8;
      int nn = n0 + nt * 16 + nr;
#pragma unroll
      for (int r = 0; r < 8; ++r) {
        size_t off = (size_t)(mb + r) * N + nn;
        if (F32OUT) Cf[off] = c[r];
        else        Cb[off] = f2bf(c[r]);
      }
    }
  }
}

// ---------------------------------------------------------------------------
// Kernel 4: QKV epilogue — RoPE Q and K_new, scatter into Qbuf / caches.
// ---------------------------------------------------------------------------
__global__ void k_qkv_epilogue(const unsigned short* __restrict__ qkv,
                               unsigned short* __restrict__ Qb,
                               unsigned short* __restrict__ Kc,
                               unsigned short* __restrict__ Vc) {
  size_t idx = (size_t)blockIdx.x * blockDim.x + threadIdx.x;
  const size_t total = (size_t)B_ * Q_ * HKV_ * (G_ + 2) * 64;
  if (idx >= total) return;
  int d0 = (int)(idx & 63);
  size_t r = idx >> 6;
  int s  = (int)(r & 3);  r >>= 2;
  int kh = (int)(r & 7);  r >>= 3;
  int q  = (int)(r % Q_);
  int b  = (int)(r / Q_);

  size_t m = (size_t)b * Q_ + q;
  size_t src = m * NQKV_ + (size_t)kh * ((G_ + 2) * D_) + (size_t)s * D_ + d0;
  float x0 = bf2f(qkv[src]), x1 = bf2f(qkv[src + 64]);

  int pos = VLM_ + q;  // pos_q == position_ids[q] + KV - Q == cache slot
  float f = __powf(10000.0f, -(float)d0 * (1.0f / 64.0f));
  float sn, cs;
  __sincosf((float)pos * f, &sn, &cs);
  float y0 = x0 * cs - x1 * sn;
  float y1 = x1 * cs + x0 * sn;

  if (s < G_) {
    int h = kh * G_ + s;
    size_t dst = (((size_t)(b * H_ + h)) * Q_ + q) * D_ + d0;
    Qb[dst] = f2bf(y0);
    Qb[dst + 64] = f2bf(y1);
  } else if (s == G_) {
    size_t dst = (((size_t)(b * HKV_ + kh)) * KV_ + VLM_ + q) * D_ + d0;
    Kc[dst] = f2bf(y0);
    Kc[dst + 64] = f2bf(y1);
  } else {
    size_t dst = (((size_t)(b * HKV_ + kh)) * KV_ + VLM_ + q) * D_ + d0;
    Vc[dst] = f2bf(x0);
    Vc[dst + 64] = f2bf(x1);
  }
}

// ---------------------------------------------------------------------------
// Kernel 5: flash attention. One block per (b,h); 8 waves x 16 query rows.
// K/V tiles staged to LDS by the Tensor Data Mover (double buffered);
// S and O via bf16 WMMA; online softmax; P transposed through LDS.
// ---------------------------------------------------------------------------
__global__ void __launch_bounds__(256, 1) k_flash_attn(
    const unsigned short* __restrict__ Qb, const unsigned short* __restrict__ Kc,
    const unsigned short* __restrict__ Vc, unsigned short* __restrict__ Ob) {
  __shared__ unsigned short Ksh[2][32 * 128];   // 2 x 8 KB
  __shared__ unsigned short Vsh[2][32 * 128];   // 2 x 8 KB
  __shared__ unsigned short plds[8][16 * 32];   // per-wave P tile, 8 KB

  const int lane = threadIdx.x & 31;
  const int wave = threadIdx.x >> 5;
  const int half = lane >> 4;
  const int bh = blockIdx.x;
  const int b = bh / H_, h = bh % H_;
  const int kh = h / G_;

  const unsigned short* Qbase = Qb + ((size_t)(b * H_ + h)) * Q_ * D_;
  const unsigned short* Kbase = Kc + ((size_t)(b * HKV_ + kh)) * KV_ * D_;
  const unsigned short* Vbase = Vc + ((size_t)(b * HKV_ + kh)) * KV_ * D_;
  const int q0 = wave * 16;

  v16bf qf[4];
#pragma unroll
  for (int c = 0; c < 4; ++c) qf[c] = load_a_frag(Qbase, D_, q0, 32 * c, lane);

  v8f o[8];
#pragma unroll
  for (int j = 0; j < 8; ++j) o[j] = (v8f){};
  float rowmax[8], rowsum[8];
#pragma unroll
  for (int r = 0; r < 8; ++r) { rowmax[r] = -3.0e38f; rowsum[r] = 0.f; }

  unsigned short* myp = plds[wave];
  const float sc = 0.08838834764831845f;  // 1/sqrt(128)
  const int NT = KV_ / 32;                // 68 tiles

  // prologue: DMA tile 0 into buffer 0
  if (wave == 0) {
    tdm_load_tile_32x128(Kbase, (unsigned int)(uintptr_t)&Ksh[0][0]);
    tdm_load_tile_32x128(Vbase, (unsigned int)(uintptr_t)&Vsh[0][0]);
  }

  for (int t = 0; t < NT; ++t) {
    const int kv0 = t * 32;
    const int buf = t & 1;
    if (wave == 0) __builtin_amdgcn_s_wait_tensorcnt(0);  // tile t arrived
    __syncthreads();
    if (wave == 0 && t + 1 < NT) {  // DMA tile t+1 while computing tile t
      tdm_load_tile_32x128(Kbase + (size_t)(kv0 + 32) * D_,
                           (unsigned int)(uintptr_t)&Ksh[buf ^ 1][0]);
      tdm_load_tile_32x128(Vbase + (size_t)(kv0 + 32) * D_,
                           (unsigned int)(uintptr_t)&Vsh[buf ^ 1][0]);
    }

    // S = Q · K^T over this 32-wide KV tile (K fragments from LDS)
    const unsigned short* Kt = &Ksh[buf][0];
    v8f s0 = {}, s1 = {};
#pragma unroll
    for (int c = 0; c < 4; ++c) {
      v16bf kb0 = load_b_frag(Kt, D_, 0, 32 * c, lane);
      v16bf kb1 = load_b_frag(Kt, D_, 16, 32 * c, lane);
      s0 = wmma_bf16(qf[c], kb0, s0);
      s1 = wmma_bf16(qf[c], kb1, s1);
    }

    // scale + causal mask + online softmax update
#pragma unroll
    for (int r = 0; r < 8; ++r) {
      int qi = q0 + r + half * 8;
      int lim = VLM_ + qi;
      int n0 = kv0 + (lane & 15);
      float v0 = s0[r] * sc + ((n0      <= lim) ? 0.f : -1e9f);
      float v1 = s1[r] * sc + ((n0 + 16 <= lim) ? 0.f : -1e9f);
      float mx = fmaxf(v0, v1);
      mx = fmaxf(mx, __shfl_xor(mx, 1));
      mx = fmaxf(mx, __shfl_xor(mx, 2));
      mx = fmaxf(mx, __shfl_xor(mx, 4));
      mx = fmaxf(mx, __shfl_xor(mx, 8));
      float nm = fmaxf(rowmax[r], mx);
      float corr = __expf(rowmax[r] - nm);
      rowmax[r] = nm;
      float p0 = __expf(v0 - nm), p1 = __expf(v1 - nm);
      float ps = p0 + p1;
      ps += __shfl_xor(ps, 1);
      ps += __shfl_xor(ps, 2);
      ps += __shfl_xor(ps, 4);
      ps += __shfl_xor(ps, 8);
      rowsum[r] = rowsum[r] * corr + ps;
#pragma unroll
      for (int j = 0; j < 8; ++j) o[j][r] *= corr;
      // stash P (C layout) into LDS for the A-layout transpose
      int mrow = r + half * 8;
      myp[mrow * 32 + (lane & 15)]      = f2bf(p0);
      myp[mrow * 32 + 16 + (lane & 15)] = f2bf(p1);
    }
    asm volatile("s_wait_dscnt 0" ::: "memory");  // wave-private LDS RAW
    v16bf pa = load_a_frag(myp, 32, 0, 0, lane);

    // O += P · V (V fragments from LDS)
    const unsigned short* Vt = &Vsh[buf][0];
#pragma unroll
    for (int j = 0; j < 8; ++j) {
      v16bf vb = load_b_frag_kn(Vt, D_, 0, 16 * j, lane);
      o[j] = wmma_bf16(pa, vb, o[j]);
    }
  }

  // normalize and write (b, q, h*d) bf16
#pragma unroll
  for (int r = 0; r < 8; ++r) {
    float inv = 1.0f / rowsum[r];
    int m = q0 + r + half * 8;
    size_t base =
        ((size_t)(b * Q_ + m)) * (H_ * D_) + (size_t)h * D_ + (lane & 15);
#pragma unroll
    for (int j = 0; j < 8; ++j) Ob[base + j * 16] = f2bf(o[j][r] * inv);
  }
}

// ---------------------------------------------------------------------------
// Host-side orchestration
// ---------------------------------------------------------------------------
extern "C" void kernel_launch(void* const* d_in, const int* in_sizes, int n_in,
                              void* d_out, int out_size, void* d_ws,
                              size_t ws_size, hipStream_t stream) {
  (void)in_sizes; (void)n_in; (void)out_size; (void)ws_size;
  const float* hidden = (const float*)d_in[0];
  const float* vk     = (const float*)d_in[1];
  const float* vv     = (const float*)d_in[2];
  // d_in[3] position_ids / d_in[4] attention_mask are analytic (arange / causal)
  const float* wqkv   = (const float*)d_in[5];
  const float* wo     = (const float*)d_in[6];
  float* out = (float*)d_out;

  // workspace layout (bytes); total = 200 MB
  char* ws = (char*)d_ws;
  unsigned short* hiddenB = (unsigned short*)(ws + 0);            //  8 MB
  unsigned short* wqkvB   = (unsigned short*)(ws + (8u   << 20)); // 16 MB
  unsigned short* woB     = (unsigned short*)(ws + (24u  << 20)); //  8 MB
  unsigned short* qkvT    = (unsigned short*)(ws + (32u  << 20)); // 16 MB
  unsigned short* Qbuf    = (unsigned short*)(ws + (48u  << 20)); //  8 MB
  unsigned short* Kcache  = (unsigned short*)(ws + (56u  << 20)); // 68 MB
  unsigned short* Vcache  = (unsigned short*)(ws + (124u << 20)); // 68 MB
  unsigned short* attnB   = (unsigned short*)(ws + (192u << 20)); //  8 MB

  // 1) convert operands to bf16
  k_f32_to_bf16<<<4096, 256, 0, stream>>>(hidden, hiddenB, B_ * Q_ * HID_);
  k_f32_to_bf16<<<4096, 256, 0, stream>>>(wqkv, wqkvB, NQKV_ * HID_);
  k_f32_to_bf16<<<4096, 256, 0, stream>>>(wo, woB, HID_ * H_ * D_);

  // 2) RoPE + convert VLM K/V prefix into caches
  {
    size_t total = (size_t)B_ * HKV_ * VLM_ * 64;  // 16.7M
    k_rope_vlm<<<(unsigned)((total + 255) / 256), 256, 0, stream>>>(
        vk, vv, Kcache, Vcache);
  }

  // 3) QKV GEMM: (2048 x 2048) x (2048 x 4096) -> bf16 temp
  k_gemm_bf16_nt<false><<<dim3(B_ * Q_ / 64, NQKV_ / 128), 256, 0, stream>>>(
      hiddenB, wqkvB, nullptr, qkvT, B_ * Q_, NQKV_, HID_);

  // 4) RoPE Q/K_new, scatter to Qbuf + caches
  {
    size_t total = (size_t)B_ * Q_ * HKV_ * (G_ + 2) * 64;  // 4.2M
    k_qkv_epilogue<<<(unsigned)((total + 255) / 256), 256, 0, stream>>>(
        qkvT, Qbuf, Kcache, Vcache);
  }

  // 5) flash attention: one block per (b, h), TDM-staged K/V
  k_flash_attn<<<B_ * H_, 256, 0, stream>>>(Qbuf, Kcache, Vcache, attnB);

  // 6) output projection: (2048 x 2048) x (2048 x 2048) -> f32 d_out
  k_gemm_bf16_nt<true><<<dim3(B_ * Q_ / 64, HID_ / 128), 256, 0, stream>>>(
      attnB, woB, out, nullptr, B_ * Q_, HID_, H_ * D_);
}
